// Attention_24730421690884
// MI455X (gfx1250) — matrix operand order repaired and checked by reading, not compiled, not run on hardware
//
#include <hip/hip_runtime.h>
#include <hip/hip_bf16.h>
#include <stdint.h>

#define B_    512
#define N_    256
#define SD_   1024
#define HD_   512
#define AD_   256
#define EPSF  1e-8f
#define WH_LDS_STRIDE 72   // 64 bf16 row + 8 bf16 pad (4 dwords) to avoid bank conflicts

typedef __attribute__((ext_vector_type(16))) __bf16   v16bf;
typedef __attribute__((ext_vector_type(8)))  float    v8f;
typedef __attribute__((ext_vector_type(4)))  unsigned u32x4;
typedef __attribute__((ext_vector_type(8)))  int      i32x8;
typedef __attribute__((ext_vector_type(4)))  int      i32x4;

#ifndef __has_builtin
#define __has_builtin(x) 0
#endif
#if __has_builtin(__builtin_amdgcn_tensor_load_to_lds) && __has_builtin(__builtin_amdgcn_s_wait_tensorcnt)
#define USE_TDM 1
#else
#define USE_TDM 0
#endif

// ---------------------------------------------------------------- kernel 0
// Convert Wh (AD x HD, f32) to bf16 once; it then lives in L2 (256 KB).
__global__ void wh_to_bf16(const float* __restrict__ Wh, __bf16* __restrict__ whb) {
    int i = blockIdx.x * blockDim.x + threadIdx.x;   // AD_*HD_ = 131072 total
    whb[i] = (__bf16)Wh[i];
}

// ---------------------------------------------------------------- kernel 1
// sp[b][a] = dot(state[b], Ws[a]) + bs[a]   (tiny: 0.27 GFLOP, L2-resident Ws)
__global__ void sp_proj(const float* __restrict__ state, const float* __restrict__ Ws,
                        const float* __restrict__ bs, float* __restrict__ sp) {
    const int b = blockIdx.x;
    const int a = threadIdx.x;                        // 256 threads
    const float4* s = (const float4*)(state + (size_t)b * SD_);
    const float4* w = (const float4*)(Ws + (size_t)a * SD_);
    float acc = 0.f;
#pragma unroll 8
    for (int i = 0; i < SD_ / 4; ++i) {
        float4 sv = s[i], wv = w[i];
        acc = fmaf(sv.x, wv.x, acc);
        acc = fmaf(sv.y, wv.y, acc);
        acc = fmaf(sv.z, wv.z, acc);
        acc = fmaf(sv.w, wv.w, acc);
    }
    sp[(size_t)b * AD_ + a] = acc + bs[a];
}

// ---------------------------------------------------------------- kernel 2
// Fused: hp = Wh @ h + bh computed tile-wise via v_wmma_f32_16x16x32_bf16,
// reduced in-register to cosine scores (never materializing hp).
// Block = 1 batch, 16 waves; wave w owns hints rows [16w, 16w+16).
__global__ __launch_bounds__(512, 1) void scores_kernel(
        const float* __restrict__ hints, const __bf16* __restrict__ whb,
        const float* __restrict__ sp, const float* __restrict__ bh,
        float* __restrict__ scores)
{
    __shared__ __bf16 shWh[AD_ * WH_LDS_STRIDE];   // 36864 B: one K-chunk of Wh
    __shared__ float  shSp[AD_];
    __shared__ float  shBh[AD_];
    __shared__ float  shSn;

    const int tid  = threadIdx.x;
    const int b    = blockIdx.x;
    const int lane = tid & 31;
    const int wv   = tid >> 5;        // 0..15 -> n-tile
    const int nloc = lane & 15;
    const int half = lane >> 4;       // 0/1: lane half (K/M split per ISA layouts)

    // Stage sp row + bh into LDS; wave 0 computes sn = max(||sp||, eps).
    if (tid < AD_) shSp[tid] = sp[(size_t)b * AD_ + tid];
    else           shBh[tid - AD_] = bh[tid - AD_];
    __syncthreads();
    if (tid < 32) {
        float s2 = 0.f;
#pragma unroll
        for (int i = 0; i < AD_ / 32; ++i) { float v = shSp[tid + 32 * i]; s2 = fmaf(v, v, s2); }
#pragma unroll
        for (int off = 16; off > 0; off >>= 1) s2 += __shfl_down(s2, off, 32);
        if (tid == 0) shSn = fmaxf(sqrtf(s2), EPSF);
    }

    // 16 accumulator tiles: C[at] = Wh[at-tile] x hints-tile, f32 16x16 each.
    v8f C[16];
#pragma unroll
    for (int i = 0; i < 16; ++i) {
#pragma unroll
        for (int r = 0; r < 8; ++r) C[i][r] = 0.f;
    }

    // B-fragment source row: hints[b][16w + nloc][:]  (lane half h holds K 8h..8h+7 and 16+8h..16+8h+7)
    const float* hrow = hints + ((size_t)b * N_ + wv * 16 + nloc) * HD_;

    // A-fragment LDS loader: Wh tile (16 x 32 bf16), lane M = nloc,
    //   elements 0..7 -> K = half*8+j, 8..15 -> K = 16+half*8+j
    auto loadA = [&](int at, int ks, uint4* q) {
        const __bf16* wr = &shWh[(at * 16 + nloc) * WH_LDS_STRIDE + ks * 32 + half * 8];
        q[0] = *(const uint4*)(wr);
        q[1] = *(const uint4*)(wr + 16);
    };

    for (int kc = 0; kc < HD_ / 64; ++kc) {          // 8 K-chunks of 64
        __syncthreads();                             // everyone done with previous chunk
#if USE_TDM
        if (tid < 32) {
            // ---- Tensor Data Mover: 64x256 bf16 tile of Wh -> LDS with pad ----
            unsigned lds_addr = (unsigned)(size_t)&shWh[0];
            unsigned long long ga = (unsigned long long)(const void*)(whb + kc * 64);
            u32x4 g0 = { 1u,                         // count=1, user descriptor
                         lds_addr,
                         (unsigned)ga,
                         (unsigned)((ga >> 32) & 0x1FFFFFFull) | (2u << 30) }; // type=2
            i32x8 g1 = { (int)((1u << 16)            // data_size = 2 bytes
                             | (1u << 20)            // pad_enable
                             | (4u << 22)            // pad_interval: 32 dwords (128 B row)
                             | (3u << 25)),          // pad_amount: 4 dwords (8 bf16)
                         (int)(64u << 16),           // tensor_dim0 = 64 (lo16 @ d1[31:16])
                         (int)(256u << 16),          // tensor_dim1 = 256 (lo16 @ d2[31:16])
                         (int)(64u << 16),           // tile_dim0 = 64 (@ d3[31:16])
                         256,                        // tile_dim1 = 256 (@ d4[15:0])
                         HD_,                        // tensor_dim0_stride = 512 elements
                         0, 0 };
            i32x4 gz = { 0, 0, 0, 0 };
#if __clang_major__ >= 23
            i32x8 gz8 = { 0, 0, 0, 0, 0, 0, 0, 0 };
            __builtin_amdgcn_tensor_load_to_lds(g0, g1, gz, gz, gz8, 0);
#else
            __builtin_amdgcn_tensor_load_to_lds(g0, g1, gz, gz, 0);
#endif
            __builtin_amdgcn_s_wait_tensorcnt(0);
        }
#else
        {   // Fallback: cooperative copy, 2 threads per row, 64 B each
            int row  = tid >> 1;
            int part = tid & 1;
            const uint4* src = (const uint4*)(whb + (size_t)row * HD_ + kc * 64 + part * 32);
            uint4 v0 = src[0], v1 = src[1], v2 = src[2], v3 = src[3];
            uint4* dst = (uint4*)&shWh[row * WH_LDS_STRIDE + part * 32];
            dst[0] = v0; dst[1] = v1; dst[2] = v2; dst[3] = v3;
        }
#endif
        __syncthreads();

        if (kc < HD_ / 64 - 1)
            __builtin_prefetch(hrow + (kc + 1) * 64, 0, 1);

#pragma unroll
        for (int ks = 0; ks < 2; ++ks) {             // two K=32 steps per chunk
            const int kb = kc * 64 + ks * 32;
            // ---- B-fragment: hints (K x 16), elements 0..7 -> K = half*8 + j, 8..15 -> K = 16 + half*8 + j ----
            __attribute__((aligned(16))) float tf[16];
            const float* s = hrow + kb + half * 8;
            *(float4*)(tf + 0)  = *(const float4*)(s + 0);
            *(float4*)(tf + 4)  = *(const float4*)(s + 4);
            *(float4*)(tf + 8)  = *(const float4*)(s + 16);
            *(float4*)(tf + 12) = *(const float4*)(s + 20);
            v16bf Bf;
#pragma unroll
            for (int j = 0; j < 16; ++j) Bf[j] = (__bf16)tf[j];

            // ---- software-pipelined A-frags: load tile at+1 while WMMA'ing at ----
            uint4 ping[2], pong[2];
            loadA(0, ks, ping);
#pragma unroll
            for (int at = 0; at < 16; ++at) {
                uint4* cur = (at & 1) ? pong : ping;
                uint4* nxt = (at & 1) ? ping : pong;
                if (at < 15) loadA(at + 1, ks, nxt);
                union { v16bf v; uint4 q[2]; } Af;
                Af.q[0] = cur[0];
                Af.q[1] = cur[1];
                C[at] = __builtin_amdgcn_wmma_f32_16x16x32_bf16(
                            false, Af.v, false, Bf, (short)0, C[at], false, false);
            }
        }
    }

    // Reduce over a-dim: lane holds n = nloc, a = at*16 + half*8 + r.
    float hn2 = 0.f, num = 0.f;
#pragma unroll
    for (int at = 0; at < 16; ++at) {
        __attribute__((aligned(16))) float tb[8], tsp[8];
        const float* pb = &shBh[at * 16 + half * 8];
        const float* pp = &shSp[at * 16 + half * 8];
        *(float4*)(tb + 0)  = *(const float4*)(pb + 0);
        *(float4*)(tb + 4)  = *(const float4*)(pb + 4);
        *(float4*)(tsp + 0) = *(const float4*)(pp + 0);
        *(float4*)(tsp + 4) = *(const float4*)(pp + 4);
#pragma unroll
        for (int r = 0; r < 8; ++r) {
            float v = C[at][r] + tb[r];
            hn2 = fmaf(v, v, hn2);
            num = fmaf(tsp[r], v, num);
        }
    }
    hn2 += __shfl_xor(hn2, 16, 32);
    num += __shfl_xor(num, 16, 32);
    if (lane < 16) {
        float hn = fmaxf(sqrtf(hn2), EPSF);
        scores[(size_t)b * N_ + wv * 16 + nloc] = num / (shSn * hn);
    }
}

// ---------------------------------------------------------------- kernel 3
// Softmax over N fused with the weighted sum over hints (coalesced in HD).
__global__ __launch_bounds__(256, 2) void attn_out(
        const float* __restrict__ scores, const float* __restrict__ hints,
        float* __restrict__ out)
{
    __shared__ float shW[N_];
    __shared__ float red[N_];
    const int b = blockIdx.x;
    const int t = threadIdx.x;      // 256 threads

    float s = scores[(size_t)b * N_ + t];
    red[t] = s;
    __syncthreads();
#pragma unroll
    for (int off = 128; off > 0; off >>= 1) {
        if (t < off) red[t] = fmaxf(red[t], red[t + off]);
        __syncthreads();
    }
    float m = red[0];
    __syncthreads();
    float e = __expf(s - m);
    red[t] = e;
    __syncthreads();
#pragma unroll
    for (int off = 128; off > 0; off >>= 1) {
        if (t < off) red[t] += red[t + off];
        __syncthreads();
    }
    shW[t] = e * (1.f / red[0]);
    __syncthreads();

    float a0 = 0.f, a1 = 0.f;
    const float* hb = hints + (size_t)b * N_ * HD_;
    for (int n = 0; n < N_; ++n) {
        float wn = shW[n];
        a0 = fmaf(wn, hb[(size_t)n * HD_ + t],        a0);
        a1 = fmaf(wn, hb[(size_t)n * HD_ + 256 + t],  a1);
    }
    out[(size_t)b * HD_ + t]       = a0;
    out[(size_t)b * HD_ + 256 + t] = a1;
}

// ---------------------------------------------------------------- launcher
extern "C" void kernel_launch(void* const* d_in, const int* in_sizes, int n_in,
                              void* d_out, int out_size, void* d_ws, size_t ws_size,
                              hipStream_t stream) {
    const float* state = (const float*)d_in[0];
    const float* hints = (const float*)d_in[1];
    const float* Ws    = (const float*)d_in[2];
    const float* bs    = (const float*)d_in[3];
    const float* Wh    = (const float*)d_in[4];
    const float* bh    = (const float*)d_in[5];
    float* out = (float*)d_out;

    // Workspace layout: sp (512 KB) | scores (512 KB) | Wh bf16 (256 KB)
    float*  sp     = (float*)d_ws;
    float*  scoresW= sp + (size_t)B_ * AD_;
    __bf16* whb    = (__bf16*)(scoresW + (size_t)B_ * N_);

    hipLaunchKernelGGL(wh_to_bf16,    dim3((AD_ * HD_) / 256), dim3(256), 0, stream, Wh, whb);
    hipLaunchKernelGGL(sp_proj,       dim3(B_),                dim3(AD_), 0, stream, state, Ws, bs, sp);
    hipLaunchKernelGGL(scores_kernel, dim3(B_),                dim3(512), 0, stream, hints, whb, sp, bh, scoresW);
    hipLaunchKernelGGL(attn_out,      dim3(B_),                dim3(256), 0, stream, scoresW, hints, out);
}
